// AnalogRNNLayer_38354057953352
// MI455X (gfx1250) — matrix operand order repaired and checked by reading, compiled, not run on hardware
//
#include <hip/hip_runtime.h>
#include <hip/hip_bf16.h>

typedef __attribute__((ext_vector_type(16))) __bf16 v16bf;
typedef __attribute__((ext_vector_type(8)))  __bf16 v8bf;
typedef __attribute__((ext_vector_type(8)))  float  v8f;

namespace {
constexpr int kT = 1024;   // timesteps
constexpr int kB = 32;     // batch
constexpr int kI = 512;    // input dim
constexpr int kH = 512;    // hidden dim
constexpr int kG = 4 * kH; // 2048 gate rows
constexpr int kK = kI + kH;   // 1024: concatenated reduction dim of Wcat
// persistent kernel: one block per hidden tile, 2 waves = batch tiles 0/1
constexpr int kBlocksP  = 32;
constexpr int kThreadsP = 64;
// gates_x precompute GEMM
constexpr int kMT = (kT * kB) / 16;  // 2048 row tiles
// workspace layout (bytes)
constexpr size_t kOffBias  = 256;                          // 2048 f32 (8 KB)
constexpr size_t kOffHbuf  = 8448;                         // 2 x [B*H] bf16 (64 KB)
constexpr size_t kOffWcat  = 73984;                        // [2048][1024] bf16 (4 MB)
constexpr size_t kOffGx    = kOffWcat + (size_t)2048 * 1024 * 2;  // 4,268,288
constexpr size_t kGxBytes  = (size_t)kT * kB * kG * sizeof(float);
}

#define WMMA_BF16(A, Bm, C) \
  __builtin_amdgcn_wmma_f32_16x16x32_bf16(false, (A), false, (Bm), (short)0, (C), false, false)

__device__ __forceinline__ float fast_sigmoid(float x) {
  return __builtin_amdgcn_rcpf(1.0f + __builtin_amdgcn_exp2f(-1.44269504f * x));
}
__device__ __forceinline__ float fast_tanh(float x) {
  return 1.0f - 2.0f * __builtin_amdgcn_rcpf(1.0f + __builtin_amdgcn_exp2f(2.88539008f * x));
}

// A tile (16x32) from fp32 row-major [rows x 512], converting in-register.
// ISA 16-bit A layout: lane(hi=0,row=l): K = kb+0..7, kb+16..23 ; lane(hi=1): K = kb+8..15, kb+24..31
__device__ __forceinline__ v16bf load_a_f32(const float* __restrict__ src,
                                            int m0, int kb, int l, int hi) {
  const float* p0 = src + (m0 + l) * kI + kb + hi * 8;
  const float4 f0 = *(const float4*)(p0);
  const float4 f1 = *(const float4*)(p0 + 4);
  const float4 f2 = *(const float4*)(p0 + 16);
  const float4 f3 = *(const float4*)(p0 + 20);
  v16bf a;
  a[0]  = (__bf16)f0.x; a[1]  = (__bf16)f0.y; a[2]  = (__bf16)f0.z; a[3]  = (__bf16)f0.w;
  a[4]  = (__bf16)f1.x; a[5]  = (__bf16)f1.y; a[6]  = (__bf16)f1.z; a[7]  = (__bf16)f1.w;
  a[8]  = (__bf16)f2.x; a[9]  = (__bf16)f2.y; a[10] = (__bf16)f2.z; a[11] = (__bf16)f2.w;
  a[12] = (__bf16)f3.x; a[13] = (__bf16)f3.y; a[14] = (__bf16)f3.z; a[15] = (__bf16)f3.w;
  return a;
}

// A tile (16x32) from a bf16 row-major [rows x 512] matrix: no conversions, 2 x b128 loads.
__device__ __forceinline__ v16bf load_a_bf16(const __bf16* __restrict__ src,
                                             int m0, int kb, int l, int hi) {
  const __bf16* p0 = src + (m0 + l) * kH + kb + hi * 8;
  const v8bf a0 = *(const v8bf*)(p0);
  const v8bf a1 = *(const v8bf*)(p0 + 16);
  v16bf a;
#pragma unroll
  for (int i = 0; i < 8; ++i) { a[i] = a0[i]; a[8 + i] = a1[i]; }
  return a;
}

// B tile (32x16): B[k][n] = W[row n][kcol+k]. ISA B layout: lane n (hi=0): K 0..7, 16..23;
// lane n+16 (hi=1): K 8..15, 24..31.
__device__ __forceinline__ v16bf load_b_rows(const __bf16* __restrict__ base,
                                             int row, int stride, int kcol, int hi) {
  const __bf16* p = base + (size_t)row * stride + kcol + hi * 8;
  const v8bf b0 = *(const v8bf*)(p);
  const v8bf b1 = *(const v8bf*)(p + 16);
  v16bf b;
#pragma unroll
  for (int i = 0; i < 8; ++i) { b[i] = b0[i]; b[8 + i] = b1[i]; }
  return b;
}

// Prep: Wcat[g] = [W_ih[g] | W_hh[g]] bf16; bias = b_ih + b_hh; stage h0 in bf16 (buffer 1);
// zero the device barrier counter.
__global__ void lstm_prep_kernel(const float* __restrict__ Wih, const float* __restrict__ Whh,
                                 const float* __restrict__ bih, const float* __restrict__ bhh,
                                 const float* __restrict__ h0,
                                 __bf16* __restrict__ Wcat, float* __restrict__ bias,
                                 __bf16* __restrict__ hbuf, unsigned* __restrict__ barrier_cnt) {
  const int g = blockIdx.x;
  const float* si = Wih + (size_t)g * kI;
  const float* sh = Whh + (size_t)g * kH;
  __bf16* dst = Wcat + (size_t)g * kK;
  for (int c = threadIdx.x; c < kI; c += blockDim.x) {
    dst[c]      = (__bf16)si[c];
    dst[kI + c] = (__bf16)sh[c];
  }
  if (g < (kB * kH) / 256) {  // stage h0 into parity buffer 1 (read by step t=0)
    const int idx = g * 256 + threadIdx.x;
    hbuf[kB * kH + idx] = (__bf16)h0[idx];
  }
  if (threadIdx.x == 0) bias[g] = bih[g] + bhh[g];
  if (g == 0 && threadIdx.x == 0) *barrier_cnt = 0u;
}

// Fully parallel input projection: gx[t*B+b][g] = x . W_ih^T + (b_ih + b_hh)
__global__ void __launch_bounds__(256) lstm_gates_x_kernel(
    const float* __restrict__ x, const __bf16* __restrict__ Wcat,
    const float* __restrict__ bias, float* __restrict__ gx) {
  const int lane = threadIdx.x & 31;
  const int wave = threadIdx.x >> 5;
  const int w  = blockIdx.x * 8 + wave;   // 0..65535
  const int mt = w >> 5;                  // 16-row tile of T*B
  const int jt = w & 31;                  // hidden tile
  const int m0 = mt * 16;
  const int j0 = jt * 16;
  const int l  = lane & 15;
  const int hi = lane >> 4;
  const int col = j0 + l;

  v8f ai = {}, af = {}, ag = {}, ao = {};
#pragma unroll 4
  for (int kb = 0; kb < kI; kb += 32) {
    const v16bf a  = load_a_f32(x, m0, kb, l, hi);
    const v16bf w0 = load_b_rows(Wcat, 0 * kH + j0 + l, kK, kb, hi);
    const v16bf w1 = load_b_rows(Wcat, 1 * kH + j0 + l, kK, kb, hi);
    const v16bf w2 = load_b_rows(Wcat, 2 * kH + j0 + l, kK, kb, hi);
    const v16bf w3 = load_b_rows(Wcat, 3 * kH + j0 + l, kK, kb, hi);
    ai = WMMA_BF16(a, w0, ai);
    af = WMMA_BF16(a, w1, af);
    ag = WMMA_BF16(a, w2, ag);
    ao = WMMA_BF16(a, w3, ao);
  }
  const float bi_c = bias[0 * kH + col];
  const float bf_c = bias[1 * kH + col];
  const float bg_c = bias[2 * kH + col];
  const float bo_c = bias[3 * kH + col];
#pragma unroll
  for (int r = 0; r < 8; ++r) {
    float* o = gx + (size_t)(m0 + r + hi * 8) * kG;
    o[0 * kH + col] = ai[r] + bi_c;
    o[1 * kH + col] = af[r] + bf_c;
    o[2 * kH + col] = ag[r] + bg_c;
    o[3 * kH + col] = ao[r] + bo_c;
  }
}

// Persistent recurrence. USE_GX: accumulators init from precomputed gates_x (bias folded,
// preloaded for t+1 BEFORE the barrier so L2 latency overlaps the spin); serial path is the
// recurrent GEMM only, reading bf16-staged h (no conversions on the dependent chain).
template <bool USE_GX>
__global__ void __launch_bounds__(kThreadsP, 1) lstm_persistent_kernel(
    const float* __restrict__ x,  const float* __restrict__ h0,
    const float* __restrict__ c0, const __bf16* __restrict__ Wcat,
    const float* __restrict__ bias, const float* __restrict__ gx,
    __bf16* __restrict__ hbuf, unsigned* __restrict__ barrier_cnt,
    float* __restrict__ out, float* __restrict__ hf, float* __restrict__ cf) {
  const int tid  = threadIdx.x;
  const int lane = tid & 31;
  const int m_tile = tid >> 5;           // wave 0/1 = batch rows 0..15 / 16..31
  const int j_tile = blockIdx.x;         // hidden tile 0..31
  const int m0 = m_tile * 16;
  const int j0 = j_tile * 16;
  const int l  = lane & 15;
  const int hi = lane >> 4;
  const int col = j0 + l;

  // This block's recurrent weights (4 gates x 16 rows x 512 bf16 = 64 KB) -> LDS.
  __shared__ __align__(16) __bf16 smem_w[64 * kH];
  for (int e8 = tid; e8 < (64 * kH) / 8; e8 += kThreadsP) {
    const int row = e8 >> 6;             // 0..63 = gate*16 + rloc
    const int c8  = (e8 & 63) * 8;
    const int grow = (row >> 4) * kH + j0 + (row & 15);
    *(v8bf*)(smem_w + row * kH + c8) =
        *(const v8bf*)(Wcat + (size_t)grow * kK + kI + c8);
  }
  __syncthreads();

  float bi_c = 0.f, bf_c = 0.f, bg_c = 0.f, bo_c = 0.f;
  if (!USE_GX) {
    bi_c = bias[0 * kH + col];
    bf_c = bias[1 * kH + col];
    bg_c = bias[2 * kH + col];
    bo_c = bias[3 * kH + col];
  }

  float c[8];
#pragma unroll
  for (int r = 0; r < 8; ++r) c[r] = c0[(m0 + r + hi * 8) * kH + col];

  // Software-pipelined gates_x accumulator preload (step 0).
  v8f ai = {}, af = {}, ag = {}, ao = {};
  if (USE_GX) {
#pragma unroll
    for (int r = 0; r < 8; ++r) {
      const float* o = gx + (size_t)(m0 + r + hi * 8) * kG;
      ai[r] = o[0 * kH + col];
      af[r] = o[1 * kH + col];
      ag[r] = o[2 * kH + col];
      ao[r] = o[3 * kH + col];
    }
  }

#pragma unroll 1
  for (int t = 0; t < kT; ++t) {
    if (!USE_GX) {
      ai = v8f{}; af = v8f{}; ag = v8f{}; ao = v8f{};
      const float* xsrc = x + (size_t)t * kB * kI;
#pragma unroll 4
      for (int kb = 0; kb < kI; kb += 32) {
        const v16bf a  = load_a_f32(xsrc, m0, kb, l, hi);
        const v16bf w0 = load_b_rows(Wcat, 0 * kH + j0 + l, kK, kb, hi);
        const v16bf w1 = load_b_rows(Wcat, 1 * kH + j0 + l, kK, kb, hi);
        const v16bf w2 = load_b_rows(Wcat, 2 * kH + j0 + l, kK, kb, hi);
        const v16bf w3 = load_b_rows(Wcat, 3 * kH + j0 + l, kK, kb, hi);
        ai = WMMA_BF16(a, w0, ai);
        af = WMMA_BF16(a, w1, af);
        ag = WMMA_BF16(a, w2, ag);
        ao = WMMA_BF16(a, w3, ao);
      }
    }

    // Recurrent GEMM: bf16-staged h_{t-1} (parity (t+1)&1) x LDS-resident weights.
    const __bf16* hsrc = hbuf + (size_t)((t + 1) & 1) * (kB * kH);
#pragma unroll 4
    for (int kb = 0; kb < kH; kb += 32) {
      const v16bf a  = load_a_bf16(hsrc, m0, kb, l, hi);
      const v16bf w0 = load_b_rows(smem_w, 0 * 16 + l, kH, kb, hi);
      const v16bf w1 = load_b_rows(smem_w, 1 * 16 + l, kH, kb, hi);
      const v16bf w2 = load_b_rows(smem_w, 2 * 16 + l, kH, kb, hi);
      const v16bf w3 = load_b_rows(smem_w, 3 * 16 + l, kH, kb, hi);
      ai = WMMA_BF16(a, w0, ai);
      af = WMMA_BF16(a, w1, af);
      ag = WMMA_BF16(a, w2, ag);
      ao = WMMA_BF16(a, w3, ao);
    }

    // Fused LSTM elementwise: h_t -> d_out (fp32) and bf16 staging (parity t&1).
    float* out_t = out + (size_t)t * kB * kH;
    __bf16* hdst = hbuf + (size_t)(t & 1) * (kB * kH);
    const bool last = (t == kT - 1);
#pragma unroll
    for (int r = 0; r < 8; ++r) {
      const int row = m0 + r + hi * 8;
      const float gi = fast_sigmoid(ai[r] + bi_c);
      const float gf = fast_sigmoid(af[r] + bf_c);
      const float gg = fast_tanh   (ag[r] + bg_c);
      const float go = fast_sigmoid(ao[r] + bo_c);
      const float cn = gf * c[r] + gi * gg;
      const float hn = go * fast_tanh(cn);
      c[r] = cn;
      out_t[row * kH + col] = hn;
      hdst[row * kH + col]  = (__bf16)hn;
      if (last) {
        hf[row * kH + col] = hn;
        cf[row * kH + col] = cn;
      }
    }

    if (!last) {
      if (USE_GX) {
        // Preload next step's gates_x BEFORE the barrier: independent of h, so its L2
        // latency is hidden behind the release fence + spin.
        const float* g1 = gx + (size_t)(t + 1) * kB * kG;
#pragma unroll
        for (int r = 0; r < 8; ++r) {
          const float* o = g1 + (size_t)(m0 + r + hi * 8) * kG;
          ai[r] = o[0 * kH + col];
          af[r] = o[1 * kH + col];
          ag[r] = o[2 * kH + col];
          ao[r] = o[3 * kH + col];
        }
      }
      // Device-wide generation barrier (32 co-resident blocks, monotonic counter).
      __syncthreads();
      if (tid == 0) {
        __threadfence();
        __hip_atomic_fetch_add(barrier_cnt, 1u, __ATOMIC_RELEASE, __HIP_MEMORY_SCOPE_AGENT);
        const unsigned target = (unsigned)(kBlocksP * (t + 1));
        while (__hip_atomic_load(barrier_cnt, __ATOMIC_ACQUIRE, __HIP_MEMORY_SCOPE_AGENT) < target) {
          __builtin_amdgcn_s_sleep(2);
        }
      }
      __syncthreads();
      __threadfence();  // acquire visibility of other blocks' staged h_t
    }
  }
}

extern "C" void kernel_launch(void* const* d_in, const int* in_sizes, int n_in,
                              void* d_out, int out_size, void* d_ws, size_t ws_size,
                              hipStream_t stream) {
  (void)in_sizes; (void)n_in; (void)out_size;
  const float* x   = (const float*)d_in[0];  // [T,B,I]
  const float* h0  = (const float*)d_in[1];  // [B,H]
  const float* c0  = (const float*)d_in[2];  // [B,H]
  const float* Wih = (const float*)d_in[3];  // [4H,I]
  const float* Whh = (const float*)d_in[4];  // [4H,H]
  const float* bih = (const float*)d_in[5];  // [4H]
  const float* bhh = (const float*)d_in[6];  // [4H]

  float* out = (float*)d_out;                   // outputs [T,B,H]
  float* hf  = out + (size_t)kT * kB * kH;      // h_f [B,H]
  float* cf  = hf + (size_t)kB * kH;            // c_f [B,H]

  char* ws = (char*)d_ws;
  unsigned* barrier_cnt = (unsigned*)ws;
  float*    bias        = (float*)(ws + kOffBias);
  __bf16*   hbuf        = (__bf16*)(ws + kOffHbuf);
  __bf16*   Wcat        = (__bf16*)(ws + kOffWcat);
  float*    gx          = (float*)(ws + kOffGx);
  const bool use_gx = ws_size >= kOffGx + kGxBytes;

  lstm_prep_kernel<<<kG, 256, 0, stream>>>(Wih, Whh, bih, bhh, h0, Wcat, bias, hbuf,
                                           barrier_cnt);
  if (use_gx) {
    lstm_gates_x_kernel<<<(kMT * 32) / 8, 256, 0, stream>>>(x, Wcat, bias, gx);
    lstm_persistent_kernel<true><<<kBlocksP, kThreadsP, 0, stream>>>(
        x, h0, c0, Wcat, bias, gx, hbuf, barrier_cnt, out, hf, cf);
  } else {
    lstm_persistent_kernel<false><<<kBlocksP, kThreadsP, 0, stream>>>(
        x, h0, c0, Wcat, bias, gx, hbuf, barrier_cnt, out, hf, cf);
  }
}
